// AtomGIN_1503238553652
// MI455X (gfx1250) — matrix operand order repaired and checked by reading, not compiled, not run on hardware
//
#include <hip/hip_runtime.h>
#include <hip/hip_bf16.h>

// ---------- types ----------
typedef __attribute__((ext_vector_type(16))) __bf16 v16bf;
typedef __attribute__((ext_vector_type(8)))  float  v8f;
typedef __attribute__((ext_vector_type(4)))  unsigned u32x4;

union FragAB {
    v16bf v;
    u32x4 q[2];
};

__device__ __forceinline__ unsigned short f2bf(float f) {
    union { float f; unsigned u; } c; c.f = f;
    unsigned u = c.u;
    u += 0x7FFFu + ((u >> 16) & 1u);   // round-to-nearest-even
    return (unsigned short)(u >> 16);
}

// ---------- small elementwise kernels ----------
__global__ __launch_bounds__(256) void atom_embed_k(
    const int* __restrict__ x, const float* __restrict__ e0,
    const float* __restrict__ e1, float* __restrict__ h, int n) {
    int idx = blockIdx.x * 256 + threadIdx.x;
    if (idx >= n * 128) return;
    int i = idx >> 7, d = idx & 127;
    h[idx] = e0[x[i * 2] * 128 + d] + e1[x[i * 2 + 1] * 128 + d];
}

// aggr = h + (edge_emb0[l][4] + edge_emb1[l][0])   (self-loop contribution)
__global__ __launch_bounds__(256) void init_aggr_k(
    const float* __restrict__ h, const float* __restrict__ e0l,
    const float* __restrict__ e1l, float* __restrict__ aggr, int n) {
    int idx = blockIdx.x * 256 + threadIdx.x;
    if (idx >= n * 128) return;
    int d = idx & 127;
    aggr[idx] = h[idx] + e0l[4 * 128 + d] + e1l[0 * 128 + d];
}

// one edge per 128 threads: msg = h[src] + eemb; atomicAdd to aggr[dst]
__global__ __launch_bounds__(256) void scatter_k(
    const float* __restrict__ h, const int* __restrict__ src,
    const int* __restrict__ dst, const int* __restrict__ ea,
    const float* __restrict__ e0l, const float* __restrict__ e1l,
    float* __restrict__ aggr, int E) {
    int e = blockIdx.x * 2 + (threadIdx.x >> 7);
    if (e >= E) return;
    int d  = threadIdx.x & 127;
    int s  = src[e];
    int t  = dst[e];
    int a0 = ea[e * 2 + 0];
    int a1 = ea[e * 2 + 1];
    float m = h[(size_t)s * 128 + d] + e0l[a0 * 128 + d] + e1l[a1 * 128 + d];
    atomicAdd(&aggr[(size_t)t * 128 + d], m);
}

__global__ __launch_bounds__(256) void to_bf16_k(
    const float* __restrict__ src, unsigned short* __restrict__ dst, int count) {
    int idx = blockIdx.x * 256 + threadIdx.x;
    if (idx < count) dst[idx] = f2bf(src[idx]);
}

__global__ __launch_bounds__(256) void zero_u16_k(
    unsigned short* __restrict__ dst, int count) {
    int idx = blockIdx.x * 256 + threadIdx.x;
    if (idx < count) dst[idx] = 0;
}

// pack row-major f32 W[K x NC] into per-lane WMMA B fragments (bf16)
// fragment element j of lane L, tile (kt,nt):
//   col = nt*16 + (L&15),  k = kt*32 + (L>>4)*8 + (j&7) + (j>>3)*16
__global__ __launch_bounds__(256) void pack_w_k(
    const float* __restrict__ W, unsigned short* __restrict__ Bp, int K, int NC) {
    int idx = blockIdx.x * 256 + threadIdx.x;
    int NT = NC / 16;
    int total = (K / 32) * NT * 512;
    if (idx >= total) return;
    int j    = idx & 15;
    int lane = (idx >> 4) & 31;
    int tile = idx >> 9;
    int kt = tile / NT;
    int nt = tile % NT;
    int col = nt * 16 + (lane & 15);
    int k   = kt * 32 + (lane >> 4) * 8 + (j & 7) + (j >> 3) * 16;
    Bp[idx] = f2bf(W[(size_t)k * NC + col]);
}

// ---------- WMMA GEMMs (M pre-padded to multiple of 32: no guards) ----------
// A fragment (16-bit A, 16x32): lane<16 rows M=lane, K chunks {0..7,16..23};
// lane>=16 same rows, K chunks {8..15,24..31}.
__device__ __forceinline__ FragAB load_a_frag(
    const unsigned short* __restrict__ A, int lda, int arow, int kbase) {
    FragAB a;
    const u32x4* p = reinterpret_cast<const u32x4*>(A + (size_t)arow * lda + kbase);
    a.q[0] = p[0];   // K .. K+7
    a.q[1] = p[2];   // K+16 .. K+23
    return a;
}

// hid = relu(A[Mp,128] @ W1pack + b1)  -> bf16 [Mp,256]
__global__ __launch_bounds__(256) void gemm1_k(
    const unsigned short* __restrict__ A, const unsigned short* __restrict__ Bp,
    const float* __restrict__ bias, unsigned short* __restrict__ C) {
    const int K = 128, NC = 256, KT = K / 32, NT = NC / 16;
    int lane = threadIdx.x & 31;
    int wave = threadIdx.x >> 5;
    int wm = wave >> 2, wn = wave & 3;          // 2 x 4 waves: M32 x N256 block tile
    int m0 = blockIdx.x * 32 + wm * 16;
    int arow  = m0 + (lane & 15);
    int akoff = (lane >> 4) * 8;
    v8f acc[4] = {};
#pragma unroll
    for (int kt = 0; kt < KT; ++kt) {
        FragAB a = load_a_frag(A, K, arow, kt * 32 + akoff);
#pragma unroll
        for (int t = 0; t < 4; ++t) {
            FragAB b;
            const u32x4* p = reinterpret_cast<const u32x4*>(
                Bp + (((size_t)kt * NT + (wn * 4 + t)) * 32 + lane) * 16);
            b.q[0] = p[0]; b.q[1] = p[1];
            acc[t] = __builtin_amdgcn_wmma_f32_16x16x32_bf16(
                false, a.v, false, b.v, (short)0, acc[t], false, false);
        }
    }
    int row_base = m0 + (lane >> 4) * 8;
    int col_lane = lane & 15;
#pragma unroll
    for (int t = 0; t < 4; ++t) {
        int col = (wn * 4 + t) * 16 + col_lane;
        float bb = bias[col];
#pragma unroll
        for (int v = 0; v < 8; ++v) {
            float val = acc[t][v] + bb;
            val = val > 0.f ? val : 0.f;
            C[(size_t)(row_base + v) * NC + col] = f2bf(val);
        }
    }
}

// out = hid[Mp,256] @ W2pack + b2  -> f32 [Mp,128]
__global__ __launch_bounds__(256) void gemm2_k(
    const unsigned short* __restrict__ A, const unsigned short* __restrict__ Bp,
    const float* __restrict__ bias, float* __restrict__ C) {
    const int K = 256, NC = 128, KT = K / 32, NT = NC / 16;
    int lane = threadIdx.x & 31;
    int wave = threadIdx.x >> 5;
    int wm = wave >> 2, wn = wave & 3;          // 2 x 4 waves: M32 x N128 block tile
    int m0 = blockIdx.x * 32 + wm * 16;
    int arow  = m0 + (lane & 15);
    int akoff = (lane >> 4) * 8;
    v8f acc[2] = {};
#pragma unroll
    for (int kt = 0; kt < KT; ++kt) {
        FragAB a = load_a_frag(A, K, arow, kt * 32 + akoff);
#pragma unroll
        for (int t = 0; t < 2; ++t) {
            FragAB b;
            const u32x4* p = reinterpret_cast<const u32x4*>(
                Bp + (((size_t)kt * NT + (wn * 2 + t)) * 32 + lane) * 16);
            b.q[0] = p[0]; b.q[1] = p[1];
            acc[t] = __builtin_amdgcn_wmma_f32_16x16x32_bf16(
                false, a.v, false, b.v, (short)0, acc[t], false, false);
        }
    }
    int row_base = m0 + (lane >> 4) * 8;
    int col_lane = lane & 15;
#pragma unroll
    for (int t = 0; t < 2; ++t) {
        int col = (wn * 2 + t) * 16 + col_lane;
        float bb = bias[col];
#pragma unroll
        for (int v = 0; v < 8; ++v) {
            C[(size_t)(row_base + v) * NC + col] = acc[t][v] + bb;
        }
    }
}

// ---------- batchnorm ----------
__global__ void bn_zero_k(float* __restrict__ sums) {
    sums[threadIdx.x] = 0.f;   // 256 threads: 128 sums + 128 sumsq
}

__global__ __launch_bounds__(256) void bn_stats_k(
    const float* __restrict__ out, float* __restrict__ sums, int nrows) {
    __shared__ float ls[256];
    ls[threadIdx.x] = 0.f;
    __syncthreads();
    int col = threadIdx.x & 127;
    int r0 = (blockIdx.x * 256 + threadIdx.x) >> 7;
    int rstep = (gridDim.x * 256) >> 7;
    float s = 0.f, sq = 0.f;
    for (int r = r0; r < nrows; r += rstep) {
        float v = out[(size_t)r * 128 + col];
        s += v; sq += v * v;
    }
    atomicAdd(&ls[col], s);
    atomicAdd(&ls[128 + col], sq);
    __syncthreads();
    atomicAdd(&sums[threadIdx.x], ls[threadIdx.x]);
}

__global__ void bn_finalize_k(
    const float* __restrict__ sums, const float* __restrict__ gamma,
    const float* __restrict__ beta, float* __restrict__ scsh, float n) {
    int d = threadIdx.x;   // 128 threads
    float mean = sums[d] / n;
    float var  = sums[128 + d] / n - mean * mean;
    float sc = gamma[d] * rsqrtf(var + 1e-5f);
    scsh[d] = sc;
    scsh[128 + d] = beta[d] - mean * sc;
}

__global__ __launch_bounds__(256) void bn_apply_k(
    const float* __restrict__ out, const float* __restrict__ scsh,
    float* __restrict__ dst, int count, int do_relu) {
    int idx = blockIdx.x * 256 + threadIdx.x;
    if (idx >= count) return;
    int d = idx & 127;
    float v = out[idx] * scsh[d] + scsh[128 + d];
    if (do_relu) v = fmaxf(v, 0.f);
    dst[idx] = v;
}

// ---------- host launch ----------
extern "C" void kernel_launch(void* const* d_in, const int* in_sizes, int n_in,
                              void* d_out, int out_size, void* d_ws, size_t ws_size,
                              hipStream_t stream) {
    const int N  = in_sizes[0] / 2;
    const int E  = in_sizes[1] / 2;
    const int Mp = (N + 31) & ~31;      // M padded to WMMA block-tile multiple
    const int D  = 128;
    const int L  = 3;

    const int*   x     = (const int*)d_in[0];
    const int*   ei    = (const int*)d_in[1];
    const int*   ea    = (const int*)d_in[2];
    const float* ae0   = (const float*)d_in[3];
    const float* ae1   = (const float*)d_in[4];
    const float* ee0   = (const float*)d_in[5];   // [L,5,D]
    const float* ee1   = (const float*)d_in[6];   // [L,3,D]
    const float* W1    = (const float*)d_in[7];   // [L,D,2D]
    const float* b1    = (const float*)d_in[8];   // [L,2D]
    const float* W2    = (const float*)d_in[9];   // [L,2D,D]
    const float* b2    = (const float*)d_in[10];  // [L,D]
    const float* gamma = (const float*)d_in[11];
    const float* beta  = (const float*)d_in[12];

    char* ws = (char*)d_ws;
    size_t off = 0;
    auto alloc = [&](size_t bytes) -> void* {
        void* p = ws + off;
        off += (bytes + 255) & ~size_t(255);
        return p;
    };
    float*          h    = (float*)alloc((size_t)Mp * D * 4);
    float*          aggr = (float*)alloc((size_t)Mp * D * 4);   // reused as 'out'
    unsigned short* abf  = (unsigned short*)alloc((size_t)Mp * D * 2);
    unsigned short* hid  = (unsigned short*)alloc((size_t)Mp * 2 * D * 2);
    unsigned short* w1p  = (unsigned short*)alloc((size_t)L * D * 2 * D * 2);
    unsigned short* w2p  = (unsigned short*)alloc((size_t)L * 2 * D * D * 2);
    float*          bns  = (float*)alloc(2 * D * 4);
    float*          scsh = (float*)alloc(2 * D * 4);
    (void)ws_size; (void)n_in; (void)out_size;

    const int nd       = N * D;
    const int blocksND = (nd + 255) / 256;
    const int packTot  = (D / 32) * (2 * D / 16) * 512;   // same count for W1 and W2 shapes
    const int packBlk  = (packTot + 255) / 256;
    const int gemmGrid = Mp / 32;
    const int padA     = (Mp - N) * D;                    // pad rows of abf (A of gemm1)

    // zero the A-matrix pad rows once (keeps WMMA inner loops guard-free)
    if (padA > 0)
        zero_u16_k<<<(padA + 255) / 256, 256, 0, stream>>>(abf + (size_t)N * D, padA);

    // pack all layer weights into WMMA B-fragment layout (bf16)
    for (int l = 0; l < L; ++l) {
        pack_w_k<<<packBlk, 256, 0, stream>>>(W1 + (size_t)l * D * 2 * D,
                                              w1p + (size_t)l * D * 2 * D, D, 2 * D);
        pack_w_k<<<packBlk, 256, 0, stream>>>(W2 + (size_t)l * 2 * D * D,
                                              w2p + (size_t)l * 2 * D * D, 2 * D, D);
    }

    // atom embedding
    atom_embed_k<<<blocksND, 256, 0, stream>>>(x, ae0, ae1, h, N);

    for (int l = 0; l < L; ++l) {
        const float* e0l = ee0 + (size_t)l * 5 * D;
        const float* e1l = ee1 + (size_t)l * 3 * D;

        // self-loop init + edge scatter (L2-resident fp32 atomics)
        init_aggr_k<<<blocksND, 256, 0, stream>>>(h, e0l, e1l, aggr, N);
        scatter_k<<<(E + 1) / 2, 256, 0, stream>>>(h, ei, ei + E, ea, e0l, e1l, aggr, E);

        // MLP via bf16 WMMA (pad rows: abf=0 -> hid=relu(b1) -> gemm2 pad out discarded)
        to_bf16_k<<<blocksND, 256, 0, stream>>>(aggr, abf, nd);
        gemm1_k<<<gemmGrid, 256, 0, stream>>>(abf, w1p + (size_t)l * D * 2 * D,
                                              b1 + (size_t)l * 2 * D, hid);
        gemm2_k<<<gemmGrid, 256, 0, stream>>>(hid, w2p + (size_t)l * 2 * D * D,
                                              b2 + (size_t)l * D, aggr);

        // batchnorm (batch statistics over the N real rows) + relu (except last layer)
        bn_zero_k<<<1, 256, 0, stream>>>(bns);
        bn_stats_k<<<512, 256, 0, stream>>>(aggr, bns, N);
        bn_finalize_k<<<1, 128, 0, stream>>>(bns, gamma + (size_t)l * D,
                                             beta + (size_t)l * D, scsh, (float)N);
        float* dst = (l == L - 1) ? (float*)d_out : h;
        bn_apply_k<<<blocksND, 256, 0, stream>>>(aggr, scsh, dst, nd, l != L - 1);
    }
}
